// GClassifier_60601988547121
// MI455X (gfx1250) — matrix operand order repaired and checked by reading, not compiled, not run on hardware
//
// CDNA5 (gfx1250) implementation of the GClassifier reference.
// Shared 3-layer HyperGAT (WMMA f16 GEMMs + float-atomic segment softmax/scatter),
// shared macro embedding, EGNN stack run twice (macro_pos1/2), head MLP,
// sigmoid(s0-s1). All dense contractions lower to v_wmma_f32_16x16x32_f16.
// B matrices pre-packed into per-lane-contiguous fragment order (2x b128 per
// fragment); A fragments are 2x b128 per row; generic GEMM processes MT row
// tiles per wave so each B fragment feeds MT WMMAs.
//
// Input order (setup_inputs dict order): x, pin_offset, edge_weight, macro_pos1,
// macro_pos2, params(flat, insertion order), edge_index(2xP), batch, macro_index,
// macro_num. Output: 4 floats.

#include <hip/hip_runtime.h>
#include <math.h>

typedef _Float16 half_t;
typedef __attribute__((ext_vector_type(16))) _Float16 v16h;
typedef __attribute__((ext_vector_type(8)))  _Float16 v8h;
typedef __attribute__((ext_vector_type(8)))  float    v8f;

#define DEV __device__ __forceinline__

#define Nn 200000
#define Pp 600000
#define Ee 200000
#define Bb 4
#define Mm 128
#define Qq (Bb*Mm*Mm)   /* 65536 pairs */
#define Rr (Bb*Mm)      /* 512 macro rows */
#define EINP 160        /* 129 padded to 5 k-tiles */
#define HPAD 288        /* 258 padded to 9 k-tiles / 18 n-tiles */

DEV int lane_id() { return (int)(threadIdx.x & 31); }
DEV float lrelu(float x, float s) { return x >= 0.f ? x : s * x; }

// ---- WMMA fragment helpers (layouts per CDNA5 ISA §7.12.2) ----
// A 16x32 f16 tile: lane holds row (lane&15); half=lane>>4 selects K subsets
// {h*8..h*8+7} and {16+h*8..16+h*8+7} -> two contiguous 16B runs.
DEV v16h load_a(const half_t* rowp) {
  const int h = lane_id() >> 4;
  v8h lo = *(const v8h*)(rowp + h * 8);
  v8h hi = *(const v8h*)(rowp + 16 + h * 8);
  return __builtin_shufflevector(lo, hi, 0,1,2,3,4,5,6,7,8,9,10,11,12,13,14,15);
}
// B tiles pre-packed as [frag][lane][16 halves] -> two b128 loads per fragment.
DEV v16h load_bp(const half_t* Bp, int frag) {
  const half_t* p = Bp + ((size_t)frag << 9) + (lane_id() << 4);
  v8h lo = *(const v8h*)p;
  v8h hi = *(const v8h*)(p + 8);
  return __builtin_shufflevector(lo, hi, 0,1,2,3,4,5,6,7,8,9,10,11,12,13,14,15);
}
DEV v8f wmma16(v16h a, v16h b, v8f c) {
  return __builtin_amdgcn_wmma_f32_16x16x32_f16(false, a, false, b, (short)0, c, false, false);
}

DEV void atomicMaxF(float* a, float v) {
  if (v >= 0.f) atomicMax((int*)a, __float_as_int(v));
  else          atomicMin((unsigned int*)a, (unsigned int)__float_as_int(v));
}

// ---------------- utility kernels ----------------
__global__ void k_fill_f32(float* p, float v, int n) {
  int i = blockIdx.x * blockDim.x + threadIdx.x;
  if (i < n) p[i] = v;
}
__global__ void k_copy_f32(float* d, const float* s, int n) {
  int i = blockIdx.x * blockDim.x + threadIdx.x;
  if (i < n) d[i] = s[i];
}
// Pack a (K x N) f32 weight (optionally two side-by-side sources: cols<n0 from
// src0, rest from src1) into f16 WMMA-B fragment order with zero padding to
// Kp x Np. dst element i: frag=i>>9, lane=(i>>4)&31, e=i&15;
// k = kt*32 + (e<8 ? (lane>>4)*8 + e : 16 + (lane>>4)*8 + (e-8)) ; c = nt*16 + (lane&15).
__global__ void k_pack_bfrag(half_t* dst, const float* src0, int n0,
                             const float* src1, int n1, int K, int Kp, int Np) {
  int i = blockIdx.x * blockDim.x + threadIdx.x;
  int ntiles = Np >> 4;
  int total = (Kp >> 5) * ntiles * 512;
  if (i >= total) return;
  int frag = i >> 9;
  int lane = (i >> 4) & 31;
  int e = i & 15;
  int kt = frag / ntiles, nt = frag % ntiles;
  int c = nt * 16 + (lane & 15);
  int k = kt * 32 + ((e < 8) ? (lane >> 4) * 8 + e : 16 + (lane >> 4) * 8 + (e - 8));
  int N = n0 + n1;
  float v = 0.f;
  if (k < K && c < N)
    v = (c < n0) ? src0[(size_t)k * n0 + c] : src1[(size_t)k * n1 + (c - n0)];
  dst[i] = (half_t)v;
}

// ---------------- generic WMMA GEMM: C = act(A@B + bias) ----------------
// A: Mrows x K (lda, f16, rows 16B-aligned), Bp: packed fragments (ntiles per
// k-tile row), out f16 (ldc). One wave = (MT*16) x (NT*16) tile: each B
// fragment load feeds MT WMMAs.
template<int MT, int NT, int ACT>
__global__ void k_wmma_gemm(const half_t* __restrict__ A, int lda,
                            const half_t* __restrict__ Bp, int ntiles, int K,
                            const float* __restrict__ bias,
                            half_t* __restrict__ C, int ldc, int Mrows) {
  int wave = (int)((blockIdx.x * blockDim.x + threadIdx.x) >> 5);
  int row0 = wave * 16 * MT;
  if (row0 >= Mrows) return;          // whole-wave exit: EXEC all-ones for WMMA
  int nt0 = blockIdx.y * NT;
  int lane = threadIdx.x & 31;
  int arow = row0 + (lane & 15);
  v8f acc[MT][NT] = {};
  for (int kb = 0; kb < K; kb += 32) {
    v16h a[MT];
#pragma unroll
    for (int m = 0; m < MT; ++m)
      a[m] = load_a(A + (size_t)(arow + m * 16) * lda + kb);
    int fbase = (kb >> 5) * ntiles + nt0;
#pragma unroll
    for (int t = 0; t < NT; ++t) {
      v16h b = load_bp(Bp, fbase + t);
#pragma unroll
      for (int m = 0; m < MT; ++m)
        acc[m][t] = wmma16(a[m], b, acc[m][t]);
    }
  }
  int col = lane & 15, rh = (lane >> 4) * 8;
#pragma unroll
  for (int m = 0; m < MT; ++m) {
#pragma unroll
    for (int t = 0; t < NT; ++t) {
      int c = (nt0 + t) * 16 + col;
      float bv = bias[c];
#pragma unroll
      for (int r = 0; r < 8; ++r) {
        float v = acc[m][t][r] + bv;
        if (ACT == 1) v = lrelu(v, 0.1f);
        C[(size_t)(row0 + m * 16 + rh + r) * ldc + c] = (half_t)v;
      }
    }
  }
}

// ---------------- HyperGAT kernels ----------------
// layer 0 (K=8) scalar: xl = pf@wl+bl, xs = pf@swr
__global__ void k_layer0(const float* __restrict__ x, const float* __restrict__ poff,
                         const int* __restrict__ srcn,
                         const float* __restrict__ wl, const float* __restrict__ bl,
                         const float* __restrict__ swr,
                         half_t* __restrict__ xl16, half_t* __restrict__ xs16, int Pn) {
  int gid = blockIdx.x * blockDim.x + threadIdx.x;
  if (gid >= Pn * 64) return;
  int p = gid >> 6, d = gid & 63;
  int sn = srcn[p];
  float pfv[8];
#pragma unroll
  for (int k = 0; k < 6; ++k) pfv[k] = x[(size_t)sn * 6 + k];
  pfv[6] = poff[(size_t)p * 2];
  pfv[7] = poff[(size_t)p * 2 + 1];
  float a1 = bl[d], a2 = 0.f;
#pragma unroll
  for (int k = 0; k < 8; ++k) {
    a1 += pfv[k] * wl[k * 64 + d];
    a2 += pfv[k] * swr[k * 64 + d];
  }
  xl16[gid] = (half_t)a1;
  xs16[gid] = (half_t)a2;
}

// layers 1,2 (K=128): per-wave 16-pin tile; A rows gathered (nf[src] || pf),
// B = packed [gat_wl | sage_wr] (128x128). xl = cols 0-63 (+bias), xs = 64-127.
__global__ void k_conv_gemm(const half_t* __restrict__ nf16, const half_t* __restrict__ pf16,
                            const int* __restrict__ srcn, const half_t* __restrict__ WLS,
                            const float* __restrict__ bias128,
                            half_t* __restrict__ xl16, half_t* __restrict__ xs16, int Pn) {
  int wave = (int)((blockIdx.x * blockDim.x + threadIdx.x) >> 5);
  int row0 = wave * 16;
  if (row0 >= Pn) return;
  int lane = threadIdx.x & 31;
  int p = row0 + (lane & 15);
  int sn = srcn[p];
  const half_t* rn = nf16 + (size_t)sn * 64;
  const half_t* rp = pf16 + (size_t)p * 64;
  v8f acc[8] = {};
#pragma unroll
  for (int kt = 0; kt < 4; ++kt) {
    const half_t* rb = (kt < 2) ? (rn + kt * 32) : (rp + (kt - 2) * 32);
    v16h a = load_a(rb);
#pragma unroll
    for (int t = 0; t < 8; ++t) {
      v16h b = load_bp(WLS, kt * 8 + t);
      acc[t] = wmma16(a, b, acc[t]);
    }
  }
  int col = lane & 15, rh = (lane >> 4) * 8;
#pragma unroll
  for (int t = 0; t < 8; ++t) {
    int c = t * 16 + col;
    float bv = bias128[c];
#pragma unroll
    for (int r = 0; r < 8; ++r) {
      int prow = row0 + rh + r;
      float v = acc[t][r] + bv;
      if (c < 64) xl16[(size_t)prow * 64 + c] = (half_t)v;
      else        xs16[(size_t)prow * 64 + (c - 64)] = (half_t)v;
    }
  }
}

// alpha[p] = sum_c lrelu(xl + ew[net]*wr + br, 0.2) * att   (wave per pin)
__global__ void k_alpha(const half_t* __restrict__ xl16, const int* __restrict__ netn,
                        const float* __restrict__ ew, const float* __restrict__ wr,
                        const float* __restrict__ br, const float* __restrict__ att,
                        float* __restrict__ alpha, int Pn) {
  int p = (int)((blockIdx.x * blockDim.x + threadIdx.x) >> 5);
  if (p >= Pn) return;
  int lane = threadIdx.x & 31;
  float w = ew[netn[p]];
  float s = 0.f;
#pragma unroll
  for (int i = 0; i < 2; ++i) {
    int c = lane * 2 + i;
    float m = (float)xl16[(size_t)p * 64 + c] + w * wr[c] + br[c];
    m = lrelu(m, 0.2f);
    s += m * att[c];
  }
  for (int off = 16; off; off >>= 1) s += __shfl_xor(s, off, 32);
  if (lane == 0) alpha[p] = s;
}

__global__ void k_segmax(const float* __restrict__ alpha, const int* __restrict__ netn,
                         float* __restrict__ amax, int Pn) {
  int p = blockIdx.x * blockDim.x + threadIdx.x;
  if (p < Pn) atomicMaxF(&amax[netn[p]], alpha[p]);
}
__global__ void k_expsum(float* __restrict__ alpha, const int* __restrict__ netn,
                         const float* __restrict__ amax, float* __restrict__ den, int Pn) {
  int p = blockIdx.x * blockDim.x + threadIdx.x;
  if (p >= Pn) return;
  int e = netn[p];
  float ex = expf(alpha[p] - amax[e]);
  alpha[p] = ex;                       // keep ex in place
  atomicAdd(&den[e], ex);
}
__global__ void k_edge_scatter(const float* __restrict__ ex, const float* __restrict__ den,
                               const int* __restrict__ netn, const half_t* __restrict__ xl16,
                               float* __restrict__ ef32, int Pn) {
  int gid = blockIdx.x * blockDim.x + threadIdx.x;
  if (gid >= Pn * 64) return;
  int p = gid >> 6, d = gid & 63;
  int e = netn[p];
  float a = ex[p] / den[e];
  atomicAdd(&ef32[(size_t)e * 64 + d], (float)xl16[gid] * a);
}
__global__ void k_edge_finish(const float* __restrict__ ef32, const float* __restrict__ gbias,
                              half_t* __restrict__ ef16, int En) {
  int gid = blockIdx.x * blockDim.x + threadIdx.x;
  if (gid >= En * 64) return;
  float v = lrelu(ef32[gid] + gbias[gid & 63], 0.1f);
  ef16[gid] = (half_t)v;
}

// pin_out = lrelu(ef[net]@sage_wl + sbl + xs); scatter to nodeS; store pf_next(+residual)
__global__ void k_pinout(const half_t* __restrict__ ef16, const half_t* __restrict__ xs16,
                         const int* __restrict__ srcn, const int* __restrict__ netn,
                         const half_t* __restrict__ SWL, const float* __restrict__ sbl,
                         const half_t* __restrict__ pf_in, half_t* __restrict__ pf_out,
                         float* __restrict__ nodeS, int Pn) {
  int wave = (int)((blockIdx.x * blockDim.x + threadIdx.x) >> 5);
  int row0 = wave * 16;
  if (row0 >= Pn) return;
  int lane = threadIdx.x & 31;
  int p = row0 + (lane & 15);
  int e = netn[p];
  const half_t* arow = ef16 + (size_t)e * 64;
  v8f acc[4] = {};
#pragma unroll
  for (int kt = 0; kt < 2; ++kt) {
    v16h a = load_a(arow + kt * 32);
#pragma unroll
    for (int t = 0; t < 4; ++t) {
      v16h b = load_bp(SWL, kt * 4 + t);
      acc[t] = wmma16(a, b, acc[t]);
    }
  }
  int col = lane & 15, rh = (lane >> 4) * 8;
  int s8[8];
#pragma unroll
  for (int r = 0; r < 8; ++r) s8[r] = srcn[row0 + rh + r];
#pragma unroll
  for (int t = 0; t < 4; ++t) {
    int c = t * 16 + col;
    float bv = sbl[c];
#pragma unroll
    for (int r = 0; r < 8; ++r) {
      int prow = row0 + rh + r;
      float v = acc[t][r] + bv + (float)xs16[(size_t)prow * 64 + c];
      v = lrelu(v, 0.1f);
      atomicAdd(&nodeS[(size_t)s8[r] * 64 + c], v);
      float res = pf_in ? (float)pf_in[(size_t)prow * 64 + c] : 0.f;
      pf_out[(size_t)prow * 64 + c] = (half_t)(v + res);
    }
  }
}

__global__ void k_count(const int* __restrict__ srcn, float* __restrict__ cnt, int Pn) {
  int p = blockIdx.x * blockDim.x + threadIdx.x;
  if (p < Pn) atomicAdd(&cnt[srcn[p]], 1.f);
}
__global__ void k_node_finish(const float* __restrict__ nodeS, const float* __restrict__ cnt,
                              const half_t* __restrict__ nf_in, half_t* __restrict__ nf_out, int Na) {
  int gid = blockIdx.x * blockDim.x + threadIdx.x;
  if (gid >= Na * 64) return;
  int n = gid >> 6;
  float v = nodeS[gid] / fmaxf(cnt[n], 1.f);
  if (nf_in) v += (float)nf_in[gid];
  nf_out[gid] = (half_t)v;
}

// macro embedding: feats0 = lrelu(nf[macro_index] @ emb_w + emb_b)
__global__ void k_emb(const half_t* __restrict__ nf16, const int* __restrict__ midx,
                      const float* __restrict__ ew, const float* __restrict__ eb,
                      float* __restrict__ f32, half_t* __restrict__ f16, int Rrows) {
  int gid = blockIdx.x * blockDim.x + threadIdx.x;
  if (gid >= Rrows * 64) return;
  int row = gid >> 6, d = gid & 63;
  int mrow = midx[row];
  float a = eb[d];
  for (int k = 0; k < 64; ++k) a += (float)nf16[(size_t)mrow * 64 + k] * ew[k * 64 + d];
  a = lrelu(a, 0.1f);
  f32[gid] = a;
  f16[gid] = (half_t)a;
}

// ---------------- EGNN kernels ----------------
__global__ void k_build_ein(const half_t* __restrict__ f16, const float* __restrict__ coors,
                            half_t* __restrict__ ein, float* __restrict__ dist32, int Q) {
  int q = blockIdx.x * blockDim.x + threadIdx.x;
  if (q >= Q) return;
  int j = q & 127;
  int bi = q >> 7;                 // b*128 + i
  int rj = ((q >> 14) << 7) + j;   // b*128 + j
  const v8h* fiv = (const v8h*)(f16 + (size_t)bi * 64);
  const v8h* fjv = (const v8h*)(f16 + (size_t)rj * 64);
  half_t* dst = ein + (size_t)q * EINP;
  v8h* dv = (v8h*)dst;
  float d = 0.f;
#pragma unroll
  for (int c = 0; c < 4; ++c) { float r = coors[bi * 4 + c] - coors[rj * 4 + c]; d += r * r; }
  dist32[q] = d;
#pragma unroll
  for (int k = 0; k < 8; ++k) dv[k] = fiv[k];
#pragma unroll
  for (int k = 0; k < 8; ++k) dv[8 + k] = fjv[k];
  dst[128] = (half_t)d;
#pragma unroll
  for (int k = 129; k < EINP; ++k) dst[k] = (half_t)0.f;
}

// m_ij *= sigmoid(m_ij @ g_w + g_b)   (wave per pair)
__global__ void k_gate(const half_t* __restrict__ m16, const float* __restrict__ gw,
                       const float* __restrict__ gb, half_t* __restrict__ m16g, int Q) {
  int q = (int)((blockIdx.x * blockDim.x + threadIdx.x) >> 5);
  if (q >= Q) return;
  int lane = threadIdx.x & 31;
  float v0 = (float)m16[(size_t)q * 64 + lane];
  float v1 = (float)m16[(size_t)q * 64 + 32 + lane];
  float s = v0 * gw[lane] + v1 * gw[32 + lane];
  for (int off = 16; off; off >>= 1) s += __shfl_xor(s, off, 32);
  float g = 1.f / (1.f + expf(-(s + gb[0])));
  m16g[(size_t)q * 64 + lane] = (half_t)(v0 * g);
  m16g[(size_t)q * 64 + 32 + lane] = (half_t)(v1 * g);
}

__global__ void k_m_i(const half_t* __restrict__ m16g, float* __restrict__ mi, int Rrows) {
  int gid = blockIdx.x * blockDim.x + threadIdx.x;
  if (gid >= Rrows * 64) return;
  int bi = gid >> 6, d = gid & 63;
  float s = 0.f;
  for (int j = 0; j < 128; ++j) s += (float)m16g[((size_t)bi * 128 + j) * 64 + d];
  mi[gid] = s;
}

// cw[q] = ch[q] @ c_w2 + c_b2  (wave per pair; ch already lrelu'd)
__global__ void k_cwdot(const half_t* __restrict__ ch16, const float* __restrict__ w2,
                        const float* __restrict__ b2, float* __restrict__ cw, int Q) {
  int q = (int)((blockIdx.x * blockDim.x + threadIdx.x) >> 5);
  if (q >= Q) return;
  int lane = threadIdx.x & 31;
  float s = 0.f;
#pragma unroll
  for (int i = 0; i < 8; ++i) {
    int c = lane + 32 * i;
    s += (float)ch16[(size_t)q * 256 + c] * w2[c];
  }
  for (int off = 16; off; off >>= 1) s += __shfl_xor(s, off, 32);
  if (lane == 0) cw[q] = s + b2[0];
}

__global__ void k_coors(const float* __restrict__ cin, const float* __restrict__ dist32,
                        const float* __restrict__ cw, const float* __restrict__ scale_p,
                        float* __restrict__ cout, int Rrows) {
  int gid = blockIdx.x * blockDim.x + threadIdx.x;
  if (gid >= Rrows * 4) return;
  int bi = gid >> 2, c = gid & 3;
  int base = (bi >> 7) << 7;
  float sc = scale_p[0];
  float ci = cin[bi * 4 + c];
  float s = 0.f;
  for (int j = 0; j < 128; ++j) {
    int q = bi * 128 + j;
    float rel = ci - cin[(base + j) * 4 + c];
    float rn = rel / fmaxf(sqrtf(dist32[q]), 1e-8f) * sc;
    s += cw[q] * rn;
  }
  cout[bi * 4 + c] = s + ci;
}

// inp = [layernorm(feats)*g+b , m_i]  (thread per macro row)
__global__ void k_ln_concat(const float* __restrict__ f32, const float* __restrict__ mi,
                            const float* __restrict__ g, const float* __restrict__ b,
                            half_t* __restrict__ inp, int Rrows) {
  int bi = blockIdx.x * blockDim.x + threadIdx.x;
  if (bi >= Rrows) return;
  const float* xr = f32 + (size_t)bi * 64;
  float m = 0.f;
  for (int d = 0; d < 64; ++d) m += xr[d];
  m *= (1.f / 64.f);
  float v = 0.f;
  for (int d = 0; d < 64; ++d) { float t = xr[d] - m; v += t * t; }
  v *= (1.f / 64.f);
  float rs = rsqrtf(v + 1e-5f);
  for (int d = 0; d < 64; ++d) inp[(size_t)bi * 128 + d] = (half_t)((xr[d] - m) * rs * g[d] + b[d]);
  for (int d = 0; d < 64; ++d) inp[(size_t)bi * 128 + 64 + d] = (half_t)mi[(size_t)bi * 64 + d];
}

__global__ void k_residual(const half_t* __restrict__ t2, const float* __restrict__ fin,
                           float* __restrict__ fo32, half_t* __restrict__ fo16, int Rrows) {
  int gid = blockIdx.x * blockDim.x + threadIdx.x;
  if (gid >= Rrows * 64) return;
  float v = (float)t2[gid] + fin[gid];
  fo32[gid] = v;
  fo16[gid] = (half_t)v;
}

__global__ void k_head(const float* __restrict__ feats, const float* __restrict__ w1,
                       const float* __restrict__ b1, const float* __restrict__ w2,
                       const float* __restrict__ b2, const float* __restrict__ w3,
                       const float* __restrict__ b3, float* __restrict__ sout) {
  int b = threadIdx.x;
  if (b >= Bb) return;
  float g[64], h1[64], h2[64];
  for (int d = 0; d < 64; ++d) {
    float s = 0.f;
    for (int m = 0; m < Mm; ++m) s += feats[((size_t)(b * Mm + m)) * 64 + d];
    g[d] = s * (1.f / (float)Mm);
  }
  for (int d = 0; d < 64; ++d) {
    float a = b1[d];
    for (int k = 0; k < 64; ++k) a += g[k] * w1[k * 64 + d];
    h1[d] = lrelu(a, 0.1f);
  }
  for (int d = 0; d < 64; ++d) {
    float a = b2[d];
    for (int k = 0; k < 64; ++k) a += h1[k] * w2[k * 64 + d];
    h2[d] = lrelu(a, 0.1f);
  }
  float o = b3[0];
  for (int k = 0; k < 64; ++k) o += h2[k] * w3[k];
  sout[b] = o;
}

__global__ void k_final(const float* __restrict__ s, float* __restrict__ out) {
  int b = threadIdx.x;
  if (b < Bb) out[b] = 1.f / (1.f + expf(-(s[b] - s[4 + b])));
}

// ---------------- host ----------------
static inline dim3 g1(long long threads) { return dim3((unsigned)((threads + 255) / 256)); }

extern "C" void kernel_launch(void* const* d_in, const int* in_sizes, int n_in,
                              void* d_out, int out_size, void* d_ws, size_t ws_size,
                              hipStream_t stream) {
  (void)in_sizes; (void)n_in; (void)out_size; (void)ws_size;
  const int BS = 256;
  const float* x        = (const float*)d_in[0];
  const float* pin_off  = (const float*)d_in[1];
  const float* ew       = (const float*)d_in[2];
  const float* mpos1    = (const float*)d_in[3];
  const float* mpos2    = (const float*)d_in[4];
  const float* PRM      = (const float*)d_in[5];
  const int*   eidx     = (const int*)d_in[6];
  const int*   midx     = (const int*)d_in[8];
  const int*   srcn     = eidx;
  const int*   netn     = eidx + Pp;
  float*       out      = (float*)d_out;

  // -------- parameter cursor (dict insertion order) --------
  size_t po = 0;
  auto take = [&](size_t n) { const float* p = PRM + po; po += n; return p; };
  struct ConvW { const float *wl,*bl,*wr,*br,*att,*bias,*swl,*sbl,*swr; };
  ConvW cv[3];
  for (int i = 0; i < 3; ++i) {
    size_t K = (i == 0) ? 8 : 128;
    cv[i].wl = take(K * 64); cv[i].bl = take(64); cv[i].wr = take(64); cv[i].br = take(64);
    cv[i].att = take(64);    cv[i].bias = take(64);
    cv[i].swl = take(64 * 64); cv[i].sbl = take(64); cv[i].swr = take(K * 64);
  }
  const float* emb_w = take(64 * 64);
  const float* emb_b = take(64);
  struct EgW { const float *e_w1,*e_b1,*e_w2,*e_b2,*g_w,*g_b,*c_w1,*c_b1,*c_w2,*c_b2,
                           *n_w1,*n_b1,*n_w2,*n_b2,*ln_g,*ln_b,*cscale; };
  EgW eg[3];
  for (int l = 0; l < 3; ++l) {
    eg[l].e_w1 = take(129 * 258); eg[l].e_b1 = take(258);
    eg[l].e_w2 = take(258 * 64);  eg[l].e_b2 = take(64);
    eg[l].g_w  = take(64);        eg[l].g_b  = take(1);
    eg[l].c_w1 = take(64 * 256);  eg[l].c_b1 = take(256);
    eg[l].c_w2 = take(256);       eg[l].c_b2 = take(1);
    eg[l].n_w1 = take(128 * 128); eg[l].n_b1 = take(128);
    eg[l].n_w2 = take(128 * 64);  eg[l].n_b2 = take(64);
    eg[l].ln_g = take(64);        eg[l].ln_b = take(64);
    eg[l].cscale = take(1);
  }
  const float* mlp_w1 = take(64 * 64); const float* mlp_b1 = take(64);
  const float* mlp_w2 = take(64 * 64); const float* mlp_b2 = take(64);
  const float* mlp_w3 = take(64);      const float* mlp_b3 = take(1);

  // -------- workspace bump allocator --------
  char* cur = (char*)d_ws;
  auto alloc  = [&](size_t bytes) { void* p = cur; cur += (bytes + 255) & ~(size_t)255; return p; };
  auto allocH = [&](size_t n) { return (half_t*)alloc(n * sizeof(half_t)); };
  auto allocF = [&](size_t n) { return (float*)alloc(n * sizeof(float)); };

  half_t* nf16a = allocH((size_t)Nn * 64); half_t* nf16b = allocH((size_t)Nn * 64);
  half_t* pf16a = allocH((size_t)Pp * 64); half_t* pf16b = allocH((size_t)Pp * 64);
  half_t* xl16  = allocH((size_t)Pp * 64); half_t* xs16  = allocH((size_t)Pp * 64);
  half_t* ef16  = allocH((size_t)Ee * 64);
  float*  alpha = allocF(Pp);
  float*  amax  = allocF(Ee);
  float*  den   = allocF(Ee);
  float*  cnt   = allocF(Nn);
  float*  ef32  = allocF((size_t)Ee * 64);
  float*  nodeS = allocF((size_t)Nn * 64);
  // packed B-fragment weights
  half_t* WLS[2]     = { allocH(4 * 8 * 512), allocH(4 * 8 * 512) };     // 128x128
  float*  bias128[2] = { allocF(128), allocF(128) };
  half_t* SWL[3]     = { allocH(2 * 4 * 512), allocH(2 * 4 * 512), allocH(2 * 4 * 512) };
  half_t *W1p[3], *W2p[3], *CW1[3], *NW1[3], *NW2[3];
  float*  b1pad[3];
  for (int l = 0; l < 3; ++l) {
    W1p[l] = allocH(5 * 18 * 512);   // 160x288
    W2p[l] = allocH(9 * 4 * 512);    // 288x64
    CW1[l] = allocH(2 * 16 * 512);   // 64x256
    NW1[l] = allocH(4 * 8 * 512);    // 128x128
    NW2[l] = allocH(4 * 4 * 512);    // 128x64
    b1pad[l] = allocF(HPAD);
  }
  float*  feats0_32 = allocF((size_t)Rr * 64); half_t* feats0_16 = allocH((size_t)Rr * 64);
  float*  fA32 = allocF((size_t)Rr * 64);      half_t* fA16 = allocH((size_t)Rr * 64);
  float*  fB32 = allocF((size_t)Rr * 64);      half_t* fB16 = allocH((size_t)Rr * 64);
  float*  coorsA = allocF((size_t)Rr * 4);     float* coorsB = allocF((size_t)Rr * 4);
  half_t* ein16  = allocH((size_t)Qq * EINP);
  float*  dist32 = allocF(Qq);
  half_t* h16buf = allocH((size_t)Qq * HPAD);
  half_t* m16    = allocH((size_t)Qq * 64);
  half_t* m16g   = allocH((size_t)Qq * 64);
  half_t* ch16   = allocH((size_t)Qq * 256);
  float*  cw32   = allocF(Qq);
  float*  mi32   = allocF((size_t)Rr * 64);
  half_t* inp16  = allocH((size_t)Rr * 128);
  half_t* t16    = allocH((size_t)Rr * 128);
  half_t* t2_16  = allocH((size_t)Rr * 64);
  float*  svals  = allocF(8);

  // -------- weight prep: pack everything into fragment order --------
  for (int i = 1; i < 3; ++i) {
    k_pack_bfrag<<<g1(4 * 8 * 512), BS, 0, stream>>>(WLS[i - 1], cv[i].wl, 64,
                                                     cv[i].swr, 64, 128, 128, 128);
    k_fill_f32<<<g1(128), BS, 0, stream>>>(bias128[i - 1], 0.f, 128);
    k_copy_f32<<<g1(64), BS, 0, stream>>>(bias128[i - 1], cv[i].bl, 64);
  }
  for (int i = 0; i < 3; ++i)
    k_pack_bfrag<<<g1(2 * 4 * 512), BS, 0, stream>>>(SWL[i], cv[i].swl, 64,
                                                     nullptr, 0, 64, 64, 64);
  for (int l = 0; l < 3; ++l) {
    k_pack_bfrag<<<g1(5 * 18 * 512), BS, 0, stream>>>(W1p[l], eg[l].e_w1, 258,
                                                      nullptr, 0, 129, EINP, HPAD);
    k_fill_f32<<<g1(HPAD), BS, 0, stream>>>(b1pad[l], 0.f, HPAD);
    k_copy_f32<<<g1(258), BS, 0, stream>>>(b1pad[l], eg[l].e_b1, 258);
    k_pack_bfrag<<<g1(9 * 4 * 512), BS, 0, stream>>>(W2p[l], eg[l].e_w2, 64,
                                                     nullptr, 0, 258, HPAD, 64);
    k_pack_bfrag<<<g1(2 * 16 * 512), BS, 0, stream>>>(CW1[l], eg[l].c_w1, 256,
                                                      nullptr, 0, 64, 64, 256);
    k_pack_bfrag<<<g1(4 * 8 * 512), BS, 0, stream>>>(NW1[l], eg[l].n_w1, 128,
                                                     nullptr, 0, 128, 128, 128);
    k_pack_bfrag<<<g1(4 * 4 * 512), BS, 0, stream>>>(NW2[l], eg[l].n_w2, 64,
                                                     nullptr, 0, 128, 128, 64);
  }

  // -------- shared GNN --------
  k_fill_f32<<<g1(Nn), BS, 0, stream>>>(cnt, 0.f, Nn);
  k_count<<<g1(Pp), BS, 0, stream>>>(srcn, cnt, Pp);

  const long long pinWaves = (long long)(Pp / 16) * 32;  // one wave per 16-pin tile
  half_t* nf_cur = nullptr; half_t* pf_cur = nullptr;
  for (int i = 0; i < 3; ++i) {
    half_t* pf_next = (i % 2 == 0) ? pf16a : pf16b;
    half_t* nf_next = (i % 2 == 0) ? nf16a : nf16b;
    if (i == 0)
      k_layer0<<<g1((long long)Pp * 64), BS, 0, stream>>>(x, pin_off, srcn, cv[0].wl, cv[0].bl,
                                                          cv[0].swr, xl16, xs16, Pp);
    else
      k_conv_gemm<<<g1(pinWaves), BS, 0, stream>>>(nf_cur, pf_cur, srcn, WLS[i - 1],
                                                   bias128[i - 1], xl16, xs16, Pp);
    k_fill_f32<<<g1(Ee), BS, 0, stream>>>(amax, -INFINITY, Ee);
    k_fill_f32<<<g1(Ee), BS, 0, stream>>>(den, 0.f, Ee);
    k_fill_f32<<<g1((long long)Ee * 64), BS, 0, stream>>>(ef32, 0.f, Ee * 64);
    k_fill_f32<<<g1((long long)Nn * 64), BS, 0, stream>>>(nodeS, 0.f, Nn * 64);
    k_alpha<<<g1((long long)Pp * 32), BS, 0, stream>>>(xl16, netn, ew, cv[i].wr, cv[i].br,
                                                       cv[i].att, alpha, Pp);
    k_segmax<<<g1(Pp), BS, 0, stream>>>(alpha, netn, amax, Pp);
    k_expsum<<<g1(Pp), BS, 0, stream>>>(alpha, netn, amax, den, Pp);
    k_edge_scatter<<<g1((long long)Pp * 64), BS, 0, stream>>>(alpha, den, netn, xl16, ef32, Pp);
    k_edge_finish<<<g1((long long)Ee * 64), BS, 0, stream>>>(ef32, cv[i].bias, ef16, Ee);
    k_pinout<<<g1(pinWaves), BS, 0, stream>>>(ef16, xs16, srcn, netn, SWL[i], cv[i].sbl,
                                              (i > 0) ? pf_cur : nullptr, pf_next, nodeS, Pp);
    k_node_finish<<<g1((long long)Nn * 64), BS, 0, stream>>>(nodeS, cnt,
                                              (i > 0) ? nf_cur : nullptr, nf_next, Nn);
    nf_cur = nf_next; pf_cur = pf_next;
  }
  k_emb<<<g1((long long)Rr * 64), BS, 0, stream>>>(nf_cur, midx, emb_w, emb_b,
                                                   feats0_32, feats0_16, Rr);

  // -------- EGNN, run twice --------
  // Q-sized GEMMs: MT=2 (2048 waves -> 256 blocks of 8 waves)
  auto egnn_run = [&](const float* mpos, float* s_out) {
    const float*  cin  = mpos;
    const float*  fin  = feats0_32;
    const half_t* fin16 = feats0_16;
    for (int l = 0; l < 3; ++l) {
      float*  cout  = (l % 2 == 0) ? coorsA : coorsB;
      float*  fo32  = (l % 2 == 0) ? fA32 : fB32;
      half_t* fo16  = (l % 2 == 0) ? fA16 : fB16;
      k_build_ein<<<g1(Qq), BS, 0, stream>>>(fin16, cin, ein16, dist32, Qq);
      k_wmma_gemm<2, 6, 1><<<dim3(256, 3), BS, 0, stream>>>(ein16, EINP, W1p[l], 18, EINP,
                                                            b1pad[l], h16buf, HPAD, Qq);
      k_wmma_gemm<2, 4, 1><<<dim3(256, 1), BS, 0, stream>>>(h16buf, HPAD, W2p[l], 4, HPAD,
                                                            eg[l].e_b2, m16, 64, Qq);
      k_gate<<<g1((long long)Qq * 32), BS, 0, stream>>>(m16, eg[l].g_w, eg[l].g_b, m16g, Qq);
      k_m_i<<<g1((long long)Rr * 64), BS, 0, stream>>>(m16g, mi32, Rr);
      k_wmma_gemm<2, 4, 1><<<dim3(256, 4), BS, 0, stream>>>(m16g, 64, CW1[l], 16, 64,
                                                            eg[l].c_b1, ch16, 256, Qq);
      k_cwdot<<<g1((long long)Qq * 32), BS, 0, stream>>>(ch16, eg[l].c_w2, eg[l].c_b2, cw32, Qq);
      k_coors<<<g1((long long)Rr * 4), BS, 0, stream>>>(cin, dist32, cw32, eg[l].cscale, cout, Rr);
      k_ln_concat<<<g1(Rr), BS, 0, stream>>>(fin, mi32, eg[l].ln_g, eg[l].ln_b, inp16, Rr);
      k_wmma_gemm<1, 8, 1><<<dim3(4, 1), BS, 0, stream>>>(inp16, 128, NW1[l], 8, 128,
                                                          eg[l].n_b1, t16, 128, Rr);
      k_wmma_gemm<1, 4, 0><<<dim3(4, 1), BS, 0, stream>>>(t16, 128, NW2[l], 4, 128,
                                                          eg[l].n_b2, t2_16, 64, Rr);
      k_residual<<<g1((long long)Rr * 64), BS, 0, stream>>>(t2_16, fin, fo32, fo16, Rr);
      cin = cout; fin = fo32; fin16 = fo16;
    }
    k_head<<<1, 64, 0, stream>>>(fin, mlp_w1, mlp_b1, mlp_w2, mlp_b2, mlp_w3, mlp_b3, s_out);
  };
  egnn_run(mpos1, svals);
  egnn_run(mpos2, svals + 4);
  k_final<<<1, 4, 0, stream>>>(svals, out);
}
